// SelfAttentionLayer_40587440947779
// MI455X (gfx1250) — compile-verified, matrix operand order checked by reading, not run on hardware
//
#include <hip/hip_runtime.h>
#include <hip/hip_bf16.h>
#include <math.h>

// ---------------------------------------------------------------------------
// Self-attention (B=32, N=1024, C=512, Dqk=64) for MI455X (gfx1250, wave32).
// All GEMMs via v_wmma_f32_16x16x32_f16. Flash-attention style (no NxN score
// tensor). Q/K row-major f16, V and weights stored K-innermost (transposed)
// so every WMMA fragment is exactly two 16-byte loads per lane.
// ---------------------------------------------------------------------------

#define B_   32
#define N_   1024
#define C_   512
#define D_   64
#define TOK  32          // tokens per qkv-projection block
#define XLD  520         // padded LDS row stride (halves): 1040 B, 16B-aligned rows

typedef __attribute__((ext_vector_type(16))) _Float16 v16h;
typedef __attribute__((ext_vector_type(8)))  _Float16 v8h;
typedef __attribute__((ext_vector_type(8)))  float    v8f;

// ---- Generic WMMA fragment loader (ISA 7.12.2 wave32 layouts) -------------
// Works for A (lane -> M) and B (lane -> N) operands when the source matrix
// is stored with the K dimension innermost (contiguous):
//   element (row, k) at p[(row0+row)*ld + k0 + k]
// Per lane the 16 halves are two contiguous 8-half runs -> two b128 loads.
// Requires ld % 8 == 0 and (row0*ld + k0) 8-half aligned (all call sites OK).
__device__ __forceinline__ v16h frag_ld(const _Float16* p, int ld,
                                        int row0, int k0) {
  const int lane = threadIdx.x & 31;
  const int m    = lane & 15;
  const int kb   = (lane >> 4) << 3;   // 0 or 8: half-wave K split
  const _Float16* base = p + (size_t)(row0 + m) * ld + k0 + kb;
  union { v16h f; v8h h[2]; } u;
  u.h[0] = *(const v8h*)(base);        // k = kb .. kb+7
  u.h[1] = *(const v8h*)(base + 16);   // k = 16+kb .. 16+kb+7
  return u.f;
}

__device__ __forceinline__ v8f wmma16(v16h a, v16h b, v8f c) {
  return __builtin_amdgcn_wmma_f32_16x16x32_f16(
      /*neg_a=*/false, a, /*neg_b=*/false, b,
      /*c_mod=*/(short)0, c, /*reuse_a=*/false, /*reuse_b=*/false);
}

// ---------------------------------------------------------------------------
// Kernel 0: one-time fp32 -> f16 weight convert + transpose to cout-major
// (K=cin innermost) so projection B-fragments are contiguous. Tiny kernel.
// ---------------------------------------------------------------------------
__global__ __launch_bounds__(256) void wcvt_kernel(
    const float* __restrict__ wq, const float* __restrict__ wk,
    const float* __restrict__ wv,
    _Float16* __restrict__ wqT, _Float16* __restrict__ wkT,
    _Float16* __restrict__ wvT) {
  const int i = blockIdx.x * blockDim.x + threadIdx.x;
  if (i < C_ * D_) {                       // [cin][cout] -> [cout][cin]
    const int cin = i / D_, cout = i % D_;
    wqT[(size_t)cout * C_ + cin] = (_Float16)wq[i];
    wkT[(size_t)cout * C_ + cin] = (_Float16)wk[i];
  }
  if (i < C_ * C_) {
    const int cin = i / C_, cout = i % C_;
    wvT[(size_t)cout * C_ + cin] = (_Float16)wv[i];
  }
}

// ---------------------------------------------------------------------------
// Kernel 1: QKV projection. Block = 32 tokens of one batch, 8 waves.
// x tile staged f16 in LDS; 80 output 16x16 tiles split across waves;
// each tile = 16 K-steps of v_wmma_f32_16x16x32_f16.
// q/k written row-major [token][d]; v written transposed [channel][token]
// (contiguous per-lane store, and attention reads it K-innermost).
// ---------------------------------------------------------------------------
__global__ __launch_bounds__(256) void qkv_proj_kernel(
    const float* __restrict__ x,
    const _Float16* __restrict__ wqT, const float* __restrict__ bq,
    const _Float16* __restrict__ wkT, const float* __restrict__ bk,
    const _Float16* __restrict__ wvT, const float* __restrict__ bv,
    _Float16* __restrict__ qf, _Float16* __restrict__ kf,
    _Float16* __restrict__ vT) {
  __shared__ _Float16 xs[TOK * XLD];   // 32 x 520 halves = 33.3 KB

  const int b  = blockIdx.x >> 5;           // N_/TOK = 32 tiles per batch
  const int t0 = (blockIdx.x & 31) * TOK;
  const float* xrow = x + ((size_t)b * N_ + t0) * C_;

  // Cooperative load + fp32->f16 convert of the 32x512 x-tile.
  for (int i = threadIdx.x * 4; i < TOK * C_; i += blockDim.x * 4) {
    const int r = i >> 9;          // / 512
    const int c = i & (C_ - 1);
    const float4 xv = *(const float4*)(xrow + (size_t)r * C_ + c);
    _Float16* dst = xs + r * XLD + c;
    dst[0] = (_Float16)xv.x; dst[1] = (_Float16)xv.y;
    dst[2] = (_Float16)xv.z; dst[3] = (_Float16)xv.w;
  }
  __syncthreads();

  const int wid  = threadIdx.x >> 5;
  const int lane = threadIdx.x & 31;
  const int col  = lane & 15;
  const int rbase = (lane >> 4) << 3;  // C/D: lanes16-31 hold rows 8..15

  // 80 output tiles: 8 for q, 8 for k, 64 for v.
  for (int t = wid; t < 80; t += 8) {
    const _Float16* wmat; const float* bias; int mt, nt;
    if (t < 8)        { wmat = wqT; bias = bq; mt = t >> 2;        nt = t & 3; }
    else if (t < 16)  { wmat = wkT; bias = bk; mt = (t - 8) >> 2;  nt = (t - 8) & 3; }
    else              { wmat = wvT; bias = bv; mt = (t - 16) >> 5; nt = (t - 16) & 31; }
    const int m0 = mt * 16, n0 = nt * 16;

    v8f acc;
    const float bval = bias[n0 + col];   // bias folded into accumulator init
#pragma unroll
    for (int r = 0; r < 8; ++r) acc[r] = bval;

#pragma unroll 4
    for (int kk = 0; kk < C_; kk += 32) {
      const v16h a  = frag_ld(xs,   XLD, m0, kk);   // ds_load_b128 x2
      const v16h bf = frag_ld(wmat, C_,  n0, kk);   // global_load_b128 x2
      acc = wmma16(a, bf, acc);
    }

    if (t < 16) {
      // q/k row-major [token][d]
      _Float16* outp = (t < 8 ? qf : kf) +
                       ((size_t)b * N_ + t0 + m0) * D_ + n0;
#pragma unroll
      for (int r = 0; r < 8; ++r)
        outp[(size_t)(rbase + r) * D_ + col] = (_Float16)acc[r];
    } else {
      // vT [channel][token]: per lane 8 contiguous halves -> b128 store
      _Float16* outp = vT + (size_t)b * C_ * N_ +
                       (size_t)(n0 + col) * N_ + (t0 + m0 + rbase);
#pragma unroll
      for (int r = 0; r < 8; ++r) outp[r] = (_Float16)acc[r];
    }
  }
}

// ---------------------------------------------------------------------------
// Kernel 2: flash attention + residual. Block = 32 query rows of one batch.
// Wave w: query row-tile (w&1), channel chunk (w>>1)*128. Online softmax;
// P re-laid out via per-wave LDS tile for the P@V WMMAs.
// ---------------------------------------------------------------------------
__global__ __launch_bounds__(256) void attn_kernel(
    const float* __restrict__ x, float* __restrict__ out,
    const _Float16* __restrict__ qf, const _Float16* __restrict__ kf,
    const _Float16* __restrict__ vT) {
  __shared__ _Float16 plds[8][16 * 32];   // 8 KB: one 16x32 P tile per wave

  const int b    = blockIdx.x >> 5;
  const int q0   = (blockIdx.x & 31) * 32;
  const int wid  = threadIdx.x >> 5;
  const int lane = threadIdx.x & 31;
  const int col  = lane & 15;
  const int rbase = (lane >> 4) << 3;
  const int mt = wid & 1;
  const int c0 = (wid >> 1) * 128;
  const int m0 = q0 + mt * 16;

  const _Float16* qb  = qf + (size_t)b * N_ * D_;
  const _Float16* kb  = kf + (size_t)b * N_ * D_;
  const _Float16* vTb = vT + (size_t)b * C_ * N_;

  // Q fragments are loop-invariant: hoist (d = 0..31, 32..63).
  const v16h aq0 = frag_ld(qb, D_, m0, 0);
  const v16h aq1 = frag_ld(qb, D_, m0, 32);

  v8f acc[8];
#pragma unroll
  for (int t = 0; t < 8; ++t)
#pragma unroll
    for (int r = 0; r < 8; ++r) acc[t][r] = 0.0f;

  float mrow[8], lrow[8];
#pragma unroll
  for (int r = 0; r < 8; ++r) { mrow[r] = -1e30f; lrow[r] = 0.0f; }

  const float scale = 0.125f;   // 1/sqrt(64)
  _Float16* myP = plds[wid];

  for (int j0 = 0; j0 < N_; j0 += 32) {
    // ---- S = q . k^T for 16 rows x 32 keys (4 WMMAs, K=64) ----
    // k rows (lane -> key, d contiguous) serve as the B operand of q.k^T.
    v8f s0 = {}, s1 = {};
    s0 = wmma16(aq0, frag_ld(kb, D_, j0,      0),  s0);
    s0 = wmma16(aq1, frag_ld(kb, D_, j0,      32), s0);
    s1 = wmma16(aq0, frag_ld(kb, D_, j0 + 16, 0),  s1);
    s1 = wmma16(aq1, frag_ld(kb, D_, j0 + 16, 32), s1);

    // Prefetch next V^T tile into cache (global_prefetch_b8).
    if (j0 + 32 < N_)
      __builtin_prefetch(vTb + (size_t)(c0 + col) * N_ + j0 + 32, 0, 1);

    // ---- online softmax; rows live per 16-lane half (C/D layout) ----
#pragma unroll
    for (int r = 0; r < 8; ++r) {
      const float a0 = s0[r] * scale;
      const float a1 = s1[r] * scale;
      float mx = fmaxf(a0, a1);
#pragma unroll
      for (int off = 8; off >= 1; off >>= 1)
        mx = fmaxf(mx, __shfl_xor(mx, off, 32));
      const float mnew = fmaxf(mrow[r], mx);
      const float corr = __expf(mrow[r] - mnew);
      const float p0 = __expf(a0 - mnew);
      const float p1 = __expf(a1 - mnew);
      float rs = p0 + p1;
#pragma unroll
      for (int off = 8; off >= 1; off >>= 1)
        rs += __shfl_xor(rs, off, 32);
      lrow[r] = lrow[r] * corr + rs;
      mrow[r] = mnew;
#pragma unroll
      for (int t = 0; t < 8; ++t) acc[t][r] *= corr;

      const int prow = rbase + r;            // C/D-layout row for this lane
      myP[prow * 32 + col]      = (_Float16)p0;
      myP[prow * 32 + 16 + col] = (_Float16)p1;
    }

    // Wave-private LDS round-trip converts P from C/D layout to A layout;
    // in-wave write->read ordering is enforced by compiler dscnt waits.
    const v16h ap = frag_ld(myP, 32, 0, 0);

    // ---- acc += P @ V over this wave's 128-channel chunk (8 WMMAs) ----
    // V^T rows (lane -> channel, token=K contiguous) are the B operand.
#pragma unroll
    for (int t = 0; t < 8; ++t) {
      const v16h bvf = frag_ld(vTb, N_, c0 + t * 16, j0);
      acc[t] = wmma16(ap, bvf, acc[t]);
    }
  }

  // ---- epilogue: normalize, add residual, store fp32 ----
#pragma unroll
  for (int r = 0; r < 8; ++r) {
    const float inv = 1.0f / lrow[r];
    const int row = m0 + rbase + r;
    const float* xr   = x   + ((size_t)b * N_ + row) * C_;
    float*       orow = out + ((size_t)b * N_ + row) * C_;
#pragma unroll
    for (int t = 0; t < 8; ++t) {
      const int cc = c0 + t * 16 + col;
      orow[cc] = acc[t][r] * inv + xr[cc];
    }
  }
}

// ---------------------------------------------------------------------------
extern "C" void kernel_launch(void* const* d_in, const int* in_sizes, int n_in,
                              void* d_out, int out_size, void* d_ws,
                              size_t ws_size, hipStream_t stream) {
  const float* x  = (const float*)d_in[0];
  const float* wq = (const float*)d_in[1];
  const float* bq = (const float*)d_in[2];
  const float* wk = (const float*)d_in[3];
  const float* bk = (const float*)d_in[4];
  const float* wv = (const float*)d_in[5];
  const float* bv = (const float*)d_in[6];
  float* out = (float*)d_out;

  // Workspace layout (halves): qf | kf | vT | wqT | wkT | wvT  (~42.6 MB)
  _Float16* ws  = (_Float16*)d_ws;
  _Float16* qf  = ws;
  _Float16* kf  = qf  + (size_t)B_ * N_ * D_;
  _Float16* vT  = kf  + (size_t)B_ * N_ * D_;
  _Float16* wqT = vT  + (size_t)B_ * N_ * C_;
  _Float16* wkT = wqT + (size_t)C_ * D_;
  _Float16* wvT = wkT + (size_t)C_ * D_;

  wcvt_kernel<<<(C_ * C_ + 255) / 256, 256, 0, stream>>>(wq, wk, wv,
                                                         wqT, wkT, wvT);
  qkv_proj_kernel<<<B_ * (N_ / TOK), 256, 0, stream>>>(
      x, wqT, bq, wkT, bk, wvT, bv, qf, kf, vT);
  attn_kernel<<<B_ * (N_ / 32), 256, 0, stream>>>(x, out, qf, kf, vT);
}
